// LorentzConv1d_15917148799336
// MI455X (gfx1250) — matrix operand order repaired and checked by reading, not compiled, not run on hardware
//
#include <hip/hip_runtime.h>

// ---------------- problem constants ----------------
#define BSZ   16
#define LLEN  8192
#define CIN   64
#define COUT  64
#define KERN  5
#define PADL  2
#define LINF  316        // (CIN-1)*KERN + 1
#define KPAD  320        // LINF padded to multiple of 32
#define KSTR  328        // LDS row stride (ushorts): 164 dwords == 36 mod 64 banks -> conflict-free A loads
#define NKT   10         // KPAD / 32
#define NNT   4          // COUT / 16
#define NFRAG (NKT*NNT)  // 40 B-matrix fragments
#define MT_PER_WAVE 2    // M-tiles per wave (B-fragment reuse)

// ---------------- WMMA types ----------------
typedef __attribute__((ext_vector_type(16))) __bf16 v16bf;
typedef __attribute__((ext_vector_type(8)))  float  v8f;

union BF16x16 {
    v16bf v;
    uint4 u[2];
};

__device__ __forceinline__ unsigned short f2bf(float f) {
    // round-to-nearest-even fp32 -> bf16
    unsigned u = __builtin_bit_cast(unsigned, f);
    u += 0x7FFFu + ((u >> 16) & 1u);
    return (unsigned short)(u >> 16);
}

// ---------------- kernel 0: pre-swizzle W into bf16 B-fragments ----------------
// B (32x16 bf16) per-lane layout as packed here (contiguous k, not the ISA table order):
//   column n = lane&15 ; lanes 0-15 hold K=0..15 (elem e -> k=e),
//   lanes 16-31 hold K=16..31 (elem e -> k=16+e).
// ws layout: frag = kt*4 + nt ; ushort index = ((frag*32 + lane)*16 + e)
__global__ void prep_w_frags(const float* __restrict__ W,
                             unsigned short* __restrict__ wf) {
    int t = blockIdx.x * blockDim.x + threadIdx.x;
    if (t >= NFRAG * 32 * 16) return;
    int e    = t & 15;
    int lane = (t >> 4) & 31;
    int frag = t >> 9;
    int nt = frag & 3;
    int kt = frag >> 2;
    int f = 32 * kt + ((lane >> 4) << 4) + e;   // feature index (K)
    int o = nt * 16 + (lane & 15);              // output channel (N)
    float v = (f < LINF) ? W[o * LINF + f] : 0.0f;
    wf[t] = f2bf(v);
}

// ---------------- kernel 1: fused feature build + WMMA GEMM + Lorentz epilogue ----------------
// 64 threads = 2 waves; each wave owns 2 consecutive M-tiles (32 output rows).
__global__ __launch_bounds__(64) void lorentz_conv_wmma(
    const float* __restrict__ x,      // (BSZ, LLEN, CIN) fp32
    const uint4* __restrict__ wfrag,  // pre-swizzled bf16 B fragments (40 KB)
    const float* __restrict__ bias,   // (COUT,)
    float*       __restrict__ out)    // (BSZ, LLEN, COUT) fp32
{
    __shared__ __align__(16) unsigned short s_feats[2 * MT_PER_WAVE][16][KSTR]; // ~41 KB

    const int tid  = threadIdx.x;
    const int wave = tid >> 5;           // 0..1
    const int lane = tid & 31;
    const int wg    = blockIdx.x;        // 2048 blocks
    const int batch = wg >> 7;           // 128 WGs per batch (512 tiles / 4)
    const int tg    = wg & 127;
    const float* xb = x + (size_t)batch * LLEN * CIN;

    // warm L2/WGP$ with the fragment buffer (global_prefetch_b8)
    if (tid < NFRAG)
        __builtin_prefetch(((const char*)wfrag) + tid * 1024, 0, 0);

    // ---------------- build features for this wave's 2 M-tiles ----------------
    for (int s = 0; s < MT_PER_WAVE; ++s) {
        const int slot = wave * MT_PER_WAVE + s;
        const int l0   = (tg * 4 + wave * MT_PER_WAVE + s) * 16;

        // zero the K padding columns (316..319): 64 entries, 2 per lane
        {
            int idx = lane * 2;
            s_feats[slot][(idx)     >> 2][LINF + ((idx)     & 3)] = 0;
            s_feats[slot][(idx + 1) >> 2][LINF + ((idx + 1) & 3)] = 0;
        }

        // space features: lane owns channels (2*lane, 2*lane+1)
        // feat f = 1 + (c-1)*5 + j comes from x[pos = l0 + m - 2 + j, c]
        {
            const int c0 = lane * 2;
            for (int p = 0; p < 16 + KERN - 1; ++p) {   // 20 window positions
                int pos = l0 - PADL + p;
                float v0 = 0.0f, v1 = 0.0f;
                if ((unsigned)pos < (unsigned)LLEN) {
                    const float* xr = xb + (size_t)pos * CIN + c0;
                    v0 = xr[0];
                    v1 = xr[1];
                }
                unsigned short b0 = f2bf(v0), b1 = f2bf(v1);
                int jlo = (p > 15) ? (p - 15) : 0;
                int jhi = (p < KERN - 1) ? p : (KERN - 1);
                for (int j = jlo; j <= jhi; ++j) {
                    int m = p - j;
                    if (c0 >= 1)   // c==0 is the time channel, handled below
                        s_feats[slot][m][1 + (c0 - 1) * KERN + j] = b0;
                    s_feats[slot][m][1 + (c0) * KERN + j] = b1;  // c = c0+1
                }
            }
        }

        // time feature: t_resc = sqrt(sum_j t_j^2 - (KERN-1))  (K_CURV=1, pad t = 1)
        if (lane < 16) {
            int m = lane;
            float acc = 0.0f;
            for (int j = 0; j < KERN; ++j) {
                int pos = l0 + m - PADL + j;
                float t = ((unsigned)pos < (unsigned)LLEN) ? xb[(size_t)pos * CIN] : 1.0f;
                acc += t * t;
            }
            s_feats[slot][m][0] = f2bf(sqrtf(fmaxf(acc - (float)(KERN - 1), 0.0f)));
        }
    }
    __syncthreads();

    // ---------------- WMMA: 2 M-tiles x 4 N-tiles, K tiled by 32 ----------------
    // A (16x32 bf16) per-lane layout: row m = lane&15;
//   lanes 0-15: elems 0..15 -> k 0..15
//   lanes 16-31: elems 0..15 -> k 16..31
    v8f acc[MT_PER_WAVE][NNT];
#pragma unroll
    for (int s = 0; s < MT_PER_WAVE; ++s)
#pragma unroll
        for (int nt = 0; nt < NNT; ++nt)
            acc[s][nt] = (v8f){};

    const int rowoff = (lane & 16) ? 16 : 0;
    const unsigned short* frow0 = &s_feats[wave * MT_PER_WAVE + 0][lane & 15][rowoff];
    const unsigned short* frow1 = &s_feats[wave * MT_PER_WAVE + 1][lane & 15][rowoff];

#pragma unroll
    for (int kt = 0; kt < NKT; ++kt) {
        BF16x16 a0, a1;
        a0.u[0] = *(const uint4*)(frow0 + 32 * kt);
        a0.u[1] = *(const uint4*)(frow0 + 32 * kt + 8);
        a1.u[0] = *(const uint4*)(frow1 + 32 * kt);
        a1.u[1] = *(const uint4*)(frow1 + 32 * kt + 8);

        const uint4* wp = wfrag + ((size_t)(kt * 4) * 32 + lane) * 2;
#pragma unroll
        for (int nt = 0; nt < NNT; ++nt) {
            BF16x16 b;
            b.u[0] = wp[nt * 64];
            b.u[1] = wp[nt * 64 + 1];
            acc[0][nt] = __builtin_amdgcn_wmma_f32_16x16x32_bf16(
                false, a0.v, false, b.v, (short)0, acc[0][nt], false, false);
            acc[1][nt] = __builtin_amdgcn_wmma_f32_16x16x32_bf16(
                false, a1.v, false, b.v, (short)0, acc[1][nt], false, false);
        }
    }

    // ---------------- epilogue: bias, Lorentz time column, store ----------------
    // C layout: lane holds column n = (lane&15) + 16*nt, rows m = v + 8*(lane>=16)
    const int nl = lane & 15;
    const float by0 = bias[nl];
    const float by1 = bias[16 + nl];
    const float by2 = bias[32 + nl];
    const float by3 = bias[48 + nl];
    const bool isTimeCol = (nl == 0);
    const int mbase = (lane >> 4) << 3;

#pragma unroll
    for (int s = 0; s < MT_PER_WAVE; ++s) {
        const int l0 = (tg * 4 + wave * MT_PER_WAVE + s) * 16;

        float ss[8];
#pragma unroll
        for (int v = 0; v < 8; ++v) {
            float y0 = acc[s][0][v] + by0;
            float y1 = acc[s][1][v] + by1;
            float y2 = acc[s][2][v] + by2;
            float y3 = acc[s][3][v] + by3;
            acc[s][0][v] = y0; acc[s][1][v] = y1;
            acc[s][2][v] = y2; acc[s][3][v] = y3;
            ss[v] = (isTimeCol ? 0.0f : y0 * y0) + y1 * y1 + y2 * y2 + y3 * y3;
        }
        // reduce across the 16 lanes of each half (rows 0-7 in lanes 0-15, rows 8-15 in 16-31)
#pragma unroll
        for (int mask = 1; mask <= 8; mask <<= 1) {
#pragma unroll
            for (int v = 0; v < 8; ++v)
                ss[v] += __shfl_xor(ss[v], mask, 32);
        }
        if (isTimeCol) {
#pragma unroll
            for (int v = 0; v < 8; ++v)
                acc[s][0][v] = sqrtf(ss[v] + 1.0f);   // y_time = sqrt(sum y_space^2 + K)
        }

#pragma unroll
        for (int v = 0; v < 8; ++v) {
            size_t row = ((size_t)batch * LLEN + (size_t)(l0 + mbase + v)) * COUT;
            out[row + nl]      = acc[s][0][v];
            out[row + 16 + nl] = acc[s][1][v];
            out[row + 32 + nl] = acc[s][2][v];
            out[row + 48 + nl] = acc[s][3][v];
        }
    }
}

// ---------------- launcher ----------------
extern "C" void kernel_launch(void* const* d_in, const int* in_sizes, int n_in,
                              void* d_out, int out_size, void* d_ws, size_t ws_size,
                              hipStream_t stream) {
    (void)in_sizes; (void)n_in; (void)out_size; (void)ws_size;
    const float* x    = (const float*)d_in[0];
    const float* W    = (const float*)d_in[1];
    const float* bias = (const float*)d_in[2];
    float* out        = (float*)d_out;

    unsigned short* wf = (unsigned short*)d_ws;  // 40 KB of bf16 B-fragments

    prep_w_frags<<<(NFRAG * 32 * 16 + 255) / 256, 256, 0, stream>>>(W, wf);

    // 8192 M-tiles total; 2 waves/WG * 2 tiles/wave = 4 tiles per WG
    lorentz_conv_wmma<<<(BSZ * LLEN / 16) / 4, 64, 0, stream>>>(
        x, (const uint4*)d_ws, bias, out);
}
